// BiDAF_52201032516068
// MI455X (gfx1250) — matrix operand order repaired and checked by reading, not compiled, not run on hardware
//
#include <hip/hip_runtime.h>

// BiDAF on MI455X (gfx1250): WMMA f16 GEMMs for all dense layers,
// TDM-staged, LDS-resident Whh WMMA recurrence for the LSTM scans.

#define BB   64     // batch
#define CC   512    // context length
#define QQ   64     // query length
#define HH   128    // hidden
#define EE   300    // embed dim
#define EPAD 320    // embed dim padded to K%32==0
#define G4   512    // 4H
#define D2   256    // 2H
#define G8   1024   // 8H

typedef _Float16 v16h __attribute__((ext_vector_type(16)));
typedef _Float16 h8   __attribute__((ext_vector_type(8)));
typedef float    v8f  __attribute__((ext_vector_type(8)));
typedef unsigned int u32x4 __attribute__((ext_vector_type(4)));
typedef int          i32x8 __attribute__((ext_vector_type(8)));

// ---------------- WMMA fragment loaders (CDNA5 16x16x32 f16 layouts) -------
// A (16x32, MxK): lane r(0..15) row M=r, K in {ko..ko+7} and {ko+16..ko+23},
// ko = 8*(lane>=16).  Two 16B loads.
__device__ __forceinline__ v16h load_a_frag(const _Float16* base, int ld, int lane) {
  int r = lane & 15, hi = lane >> 4;
  const _Float16* p = base + r * ld + hi * 8;
  union { v16h v; h8 h[2]; } u;
  u.h[0] = *(const h8*)(p);
  u.h[1] = *(const h8*)(p + 16);
  return u.v;
}
// B (32x16, KxN) from row-major W (N x K): lane r holds column n=r,
// K in {ko..ko+7} and {ko+16..ko+23}, ko = 8*(lane>=16).  Two 16B loads.
__device__ __forceinline__ v16h load_b_frag(const _Float16* base, int ld, int lane) {
  int r = lane & 15, hi = lane >> 4;
  const _Float16* p = base + r * ld + hi * 8;
  union { v16h v; h8 h[2]; } u;
  u.h[0] = *(const h8*)(p);
  u.h[1] = *(const h8*)(p + 16);
  return u.v;
}

// ---------------- generic WMMA GEMM:  D[M,N]f32 = A[M,K]f16 @ W[N,K]^T + bias[N]
// block = 256 threads = 8 waves; block tile 32(M) x 128(N): each wave owns one
// 16-col N tile and two 16-row M tiles, so every B fragment feeds two WMMAs.
__global__ void wmma_gemm_kernel(const _Float16* __restrict__ A,
                                 const _Float16* __restrict__ W,
                                 const float* __restrict__ bias,
                                 float* __restrict__ D,
                                 int M, int N, int K, int lda, int ldw, int ldd,
                                 long long sA, long long sW, long long sD,
                                 long long sBias)
{
  long long bz = blockIdx.z;
  A += bz * sA;  W += bz * sW;  D += bz * sD;
  int wave = threadIdx.x >> 5, lane = threadIdx.x & 31;
  int mBase = blockIdx.y * 32;
  int nBase = blockIdx.x * 128 + wave * 16;
  if (nBase >= N) return;                 // uniform per-wave
  v8f acc0 = {}, acc1 = {};
  const _Float16* Ab0 = A + (long long)mBase * lda;
  const _Float16* Ab1 = Ab0 + 16LL * lda;
  const _Float16* Wb  = W + (long long)nBase * ldw;
  for (int kk = 0; kk < K; kk += 32) {
    v16h a0 = load_a_frag(Ab0 + kk, lda, lane);
    v16h a1 = load_a_frag(Ab1 + kk, lda, lane);
    v16h b  = load_b_frag(Wb + kk, ldw, lane);
    acc0 = __builtin_amdgcn_wmma_f32_16x16x32_f16(false, a0, false, b,
                                                  (short)0, acc0, false, false);
    acc1 = __builtin_amdgcn_wmma_f32_16x16x32_f16(false, a1, false, b,
                                                  (short)0, acc1, false, false);
  }
  int r = lane & 15, hi = lane >> 4;
  float bv = 0.f;
  if (bias) bv = bias[bz * sBias + nBase + r];
#pragma unroll
  for (int i = 0; i < 8; ++i) {
    D[(long long)(mBase + i + 8 * hi) * ldd + nBase + r] = acc0[i] + bv;
    D[(long long)(mBase + 16 + i + 8 * hi) * ldd + nBase + r] = acc1[i] + bv;
  }
}

// ---------------- LSTM scan: Whh staged via TDM into LDS, WMMA recurrence --
// grid = (B/16, 2 dirs); block = 256 threads (8 waves).
// xg: (2, B, T, 4H) f32 precomputed x@Wih^T+b.  hs out: (B, T, 2H) f32.
#define SCAN_LDS_BYTES (G4*HH*2 + 16*G4*4 + 16*HH*4 + 16*HH*2)

// Issue one Tensor-Data-Mover load of the (4H x H) f16 Whh tile into LDS.
// D# per CDNA5 ISA: group0 = {count|flags, lds_addr,
// global_addr lo, global_addr hi | type=2}; group1 = {data_size, dims, tile
// dims, stride}.  workgroup_mask = 0 (not in a cluster).  Issued by one wave.
__device__ __forceinline__ void tdm_load_whh(const _Float16* Wd, _Float16* sWhh)
{
  unsigned ldsOff = __builtin_amdgcn_readfirstlane((unsigned)(size_t)(void*)sWhh);
  unsigned long long ga = (unsigned long long)(const void*)Wd;
  u32x4 g0;
  g0[0] = 1u;                                        // count=1 (valid user D#)
  g0[1] = ldsOff;                                    // lds_addr (bytes)
  g0[2] = (unsigned)(ga & 0xffffffffull);            // global_addr[31:0]
  g0[3] = (unsigned)((ga >> 32) & 0x01ffffffull)     // global_addr[56:32]
          | (2u << 30);                              // type = 2 ("image")
  i32x8 g1;
  g1[0] = (int)(1u << 16);                           // data_size = 1 -> 2 bytes
  g1[1] = (int)((unsigned)HH << 16);                 // tensor_dim0 = H (lo16)
  g1[2] = (int)((unsigned)G4 << 16);                 // dim0 hi16 | tensor_dim1 = 4H
  g1[3] = (int)((unsigned)HH << 16);                 // dim1 hi16 | tile_dim0 = H
  g1[4] = (int)G4;                                   // tile_dim1 = 4H, tile_dim2 = 0
  g1[5] = (int)HH;                                   // tensor_dim0_stride = H (lo32)
  g1[6] = 0;                                         // stride hi | dim1_stride lo
  g1[7] = 0;
  asm volatile("tensor_load_to_lds %0, %1" :: "s"(g0), "s"(g1) : "memory");
  __builtin_amdgcn_s_wait_tensorcnt(0);
}

__global__ void lstm_scan_kernel(const _Float16* __restrict__ Whh16, // (2,4H,H)
                                 const float* __restrict__ xg,
                                 float* __restrict__ hs,
                                 int B, int T)
{
  extern __shared__ char smemRaw[];
  _Float16* sWhh = (_Float16*)smemRaw;                       // 4H x H   (128 KB)
  float*    sGate = (float*)(smemRaw + G4 * HH * 2);         // 16 x 4H  (32 KB)
  float*    sC    = sGate + 16 * G4;                         // 16 x H   (8 KB)
  _Float16* sH    = (_Float16*)(sC + 16 * HH);               // 16 x H   (4 KB)

  int dir   = blockIdx.y;
  int bBase = blockIdx.x * 16;
  const _Float16* Wd  = Whh16 + (long long)dir * G4 * HH;
  const float*    xgd = xg + (long long)dir * B * T * G4;
  int tid = threadIdx.x;

  if (tid < 32) tdm_load_whh(Wd, sWhh);     // TDM DMA: global -> LDS (128 KB)
  for (int i = tid; i < 16 * HH; i += blockDim.x) { sC[i] = 0.f; sH[i] = (_Float16)0.f; }
  __syncthreads();

  int wave = tid >> 5, lane = tid & 31;
  int r = lane & 15, hi = lane >> 4;
  int nBase = wave * 64;                       // 8 waves cover 4H = 512

  for (int t = 0; t < T; ++t) {
    int tt = dir ? (T - 1 - t) : t;
    v8f acc[4] = {};
#pragma unroll
    for (int kk = 0; kk < HH; kk += 32) {
      v16h a = load_a_frag(sH + kk, HH, lane);
#pragma unroll
      for (int j = 0; j < 4; ++j) {
        v16h b = load_b_frag(sWhh + (nBase + j * 16) * HH + kk, HH, lane);
        acc[j] = __builtin_amdgcn_wmma_f32_16x16x32_f16(false, a, false, b,
                                                        (short)0, acc[j], false, false);
      }
    }
#pragma unroll
    for (int j = 0; j < 4; ++j) {
      int col = nBase + j * 16 + r;
#pragma unroll
      for (int i = 0; i < 8; ++i) {
        int row = i + 8 * hi;
        sGate[row * G4 + col] =
            acc[j][i] + xgd[((long long)(bBase + row) * T + tt) * G4 + col];
      }
    }
    __syncthreads();
    for (int idx = tid; idx < 16 * HH; idx += blockDim.x) {
      int row = idx >> 7, j = idx & (HH - 1);
      float ig = sGate[row * G4 + j];
      float fg = sGate[row * G4 + HH + j];
      float gg = sGate[row * G4 + 2 * HH + j];
      float og = sGate[row * G4 + 3 * HH + j];
      float iS = 1.f / (1.f + __expf(-ig));
      float fS = 1.f / (1.f + __expf(-fg));
      float gT = tanhf(gg);
      float oS = 1.f / (1.f + __expf(-og));
      float c = fS * sC[row * HH + j] + iS * gT;
      sC[row * HH + j] = c;
      float h = oS * tanhf(c);
      sH[row * HH + j] = (_Float16)h;
      hs[((long long)(bBase + row) * T + tt) * D2 + dir * HH + j] = h;
    }
    __syncthreads();
  }
}

// ---------------- small helper kernels -------------------------------------
__global__ void gather_emb_kernel(const int* __restrict__ tok,
                                  const float* __restrict__ emb,
                                  _Float16* __restrict__ dst, long long rows)
{
  long long idx = blockIdx.x * (long long)blockDim.x + threadIdx.x;
  long long tot = rows * EPAD;
  if (idx >= tot) return;
  int c = (int)(idx % EPAD);
  long long r = idx / EPAD;
  float v = (c < EE) ? emb[(long long)tok[r] * EE + c] : 0.f;
  dst[idx] = (_Float16)v;
}

// f32 -> f16 row-major cast, optional per-column scale, zero-pad K..Kpad
__global__ void cast_f16_kernel(const float* __restrict__ src,
                                _Float16* __restrict__ dst,
                                const float* __restrict__ colScale,
                                long long rows, int K, int Kpad)
{
  long long idx = blockIdx.x * (long long)blockDim.x + threadIdx.x;
  long long tot = rows * Kpad;
  if (idx >= tot) return;
  int c = (int)(idx % Kpad);
  long long r = idx / Kpad;
  float v = 0.f;
  if (c < K) {
    v = src[r * K + c];
    if (colScale) v *= colScale[c];
  }
  dst[idx] = (_Float16)v;
}

__global__ void rowdot_kernel(const float* __restrict__ X,
                              const float* __restrict__ w,
                              float* __restrict__ out, long long R, int K)
{
  long long r = blockIdx.x * (long long)blockDim.x + threadIdx.x;
  if (r >= R) return;
  const float* x = X + r * K;
  float s = 0.f;
  for (int k = 0; k < K; ++k) s += x[k] * w[k];
  out[r] = s;
}

// per (b,c) row: add rank-1 terms, softmax over Q -> f16 probs; also row max
__global__ void softmax_q_kernel(const float* __restrict__ s,
                                 const float* __restrict__ cwc,
                                 const float* __restrict__ qwq,
                                 const float* __restrict__ attb,
                                 _Float16* __restrict__ a16,
                                 float* __restrict__ smax)
{
  int row = blockIdx.x;            // b*C + c
  int b = row / CC;
  int lane = threadIdx.x;          // 32 threads (one wave32)
  const float* sr = s + (long long)row * QQ;
  float base = cwc[row] + attb[0];
  float m = -1e30f;
  for (int q = lane; q < QQ; q += 32)
    m = fmaxf(m, sr[q] + base + qwq[b * QQ + q]);
  for (int off = 16; off > 0; off >>= 1) m = fmaxf(m, __shfl_xor(m, off, 32));
  float sum = 0.f;
  for (int q = lane; q < QQ; q += 32)
    sum += __expf(sr[q] + base + qwq[b * QQ + q] - m);
  for (int off = 16; off > 0; off >>= 1) sum += __shfl_xor(sum, off, 32);
  float inv = 1.f / sum;
  for (int q = lane; q < QQ; q += 32)
    a16[(long long)row * QQ + q] =
        (_Float16)(__expf(sr[q] + base + qwq[b * QQ + q] - m) * inv);
  if (lane == 0) smax[row] = m;
}

__global__ void softmax_c_kernel(const float* __restrict__ smax,
                                 float* __restrict__ bw)
{
  int b = blockIdx.x;
  __shared__ float red[256];
  int tid = threadIdx.x;
  float m = -1e30f;
  for (int c = tid; c < CC; c += 256) m = fmaxf(m, smax[b * CC + c]);
  red[tid] = m; __syncthreads();
  for (int s = 128; s > 0; s >>= 1) { if (tid < s) red[tid] = fmaxf(red[tid], red[tid + s]); __syncthreads(); }
  m = red[0]; __syncthreads();
  float sum = 0.f;
  for (int c = tid; c < CC; c += 256) sum += __expf(smax[b * CC + c] - m);
  red[tid] = sum; __syncthreads();
  for (int s = 128; s > 0; s >>= 1) { if (tid < s) red[tid] += red[tid + s]; __syncthreads(); }
  float inv = 1.f / red[0];
  for (int c = tid; c < CC; c += 256)
    bw[b * CC + c] = __expf(smax[b * CC + c] - m) * inv;
}

__global__ void q2c_kernel(const float* __restrict__ bw,
                           const float* __restrict__ co,
                           float* __restrict__ q2c)
{
  int idx = blockIdx.x * blockDim.x + threadIdx.x;   // B * 2H
  if (idx >= BB * D2) return;
  int b = idx / D2, d = idx % D2;
  float s = 0.f;
  for (int c = 0; c < CC; ++c)
    s += bw[b * CC + c] * co[((long long)b * CC + c) * D2 + d];
  q2c[idx] = s;
}

// qoT16[b][d][q] = (f16) qo[b][q][d]   (for c2q GEMM's W operand)
__global__ void qoT_kernel(const float* __restrict__ qo, _Float16* __restrict__ qoT)
{
  long long idx = blockIdx.x * (long long)blockDim.x + threadIdx.x;
  long long tot = (long long)BB * D2 * QQ;
  if (idx >= tot) return;
  int q = (int)(idx % QQ);
  int d = (int)((idx / QQ) % D2);
  int b = (int)(idx / ((long long)QQ * D2));
  qoT[idx] = (_Float16)qo[((long long)b * QQ + q) * D2 + d];
}

// g = [co, c2q, co*c2q, co*q2c]  -> f32 (for logits) and f16 (for m1 GEMM)
__global__ void build_g_kernel(const float* __restrict__ co,
                               const float* __restrict__ c2q,
                               const float* __restrict__ q2c,
                               float* __restrict__ g32,
                               _Float16* __restrict__ g16)
{
  long long idx = blockIdx.x * (long long)blockDim.x + threadIdx.x;
  long long tot = (long long)BB * CC * D2;
  if (idx >= tot) return;
  int d = (int)(idx % D2);
  long long row = idx / D2;
  int b = (int)(row / CC);
  float c0 = co[idx];
  float cq = c2q[idx];
  float qc = q2c[b * D2 + d];
  long long o = row * G8;
  g32[o + d] = c0;              g16[o + d] = (_Float16)c0;
  g32[o + D2 + d] = cq;         g16[o + D2 + d] = (_Float16)cq;
  g32[o + 2 * D2 + d] = c0 * cq; g16[o + 2 * D2 + d] = (_Float16)(c0 * cq);
  g32[o + 3 * D2 + d] = c0 * qc; g16[o + 3 * D2 + d] = (_Float16)(c0 * qc);
}

__global__ void logits_kernel(const float* __restrict__ g,
                              const float* __restrict__ m,
                              const float* __restrict__ wg,
                              const float* __restrict__ wm,
                              const float* __restrict__ bptr,
                              float* __restrict__ out, int Kg, int Km)
{
  long long row = blockIdx.x * (long long)blockDim.x + threadIdx.x;
  if (row >= (long long)BB * CC) return;
  float s = bptr[0];
  const float* gr = g + row * Kg;
  for (int k = 0; k < Kg; ++k) s += gr[k] * wg[k];
  const float* mr = m + row * Km;
  for (int k = 0; k < Km; ++k) s += mr[k] * wm[k];
  out[row] = s;
}

__global__ void masked_softmax_kernel(const float* __restrict__ logit,
                                      const int* __restrict__ tok,
                                      float* __restrict__ out)
{
  int b = blockIdx.x;
  __shared__ float red[256];
  int tid = threadIdx.x;
  float m = -1e30f;
  for (int c = tid; c < CC; c += 256) {
    float x = (tok[b * CC + c] != 0) ? logit[b * CC + c] : -1e30f;
    m = fmaxf(m, x);
  }
  red[tid] = m; __syncthreads();
  for (int s = 128; s > 0; s >>= 1) { if (tid < s) red[tid] = fmaxf(red[tid], red[tid + s]); __syncthreads(); }
  m = red[0]; __syncthreads();
  float sum = 0.f;
  for (int c = tid; c < CC; c += 256)
    sum += (tok[b * CC + c] != 0) ? __expf(logit[b * CC + c] - m) : 0.f;
  red[tid] = sum; __syncthreads();
  for (int s = 128; s > 0; s >>= 1) { if (tid < s) red[tid] += red[tid + s]; __syncthreads(); }
  float inv = 1.f / red[0];
  for (int c = tid; c < CC; c += 256)
    out[b * CC + c] =
        ((tok[b * CC + c] != 0) ? __expf(logit[b * CC + c] - m) : 0.f) * inv;
}

// ---------------- host-side helpers ----------------------------------------
static inline unsigned cdiv(long long a, long long b) { return (unsigned)((a + b - 1) / b); }

// One BiLSTM layer: cast weights to f16, WMMA-GEMM the gate precompute for
// both directions, then run the TDM+LDS-resident WMMA scan for both directions.
static void run_bilstm(hipStream_t stream,
                       const _Float16* x16, long long rows, int T, int Kpad,
                       const float* Wih, int inDim, const float* Whh,
                       const float* bias,
                       _Float16* wih16, _Float16* whh16, float* xg,
                       float* hs_out)
{
  long long wtot = 2LL * G4 * Kpad;
  cast_f16_kernel<<<cdiv(wtot, 256), 256, 0, stream>>>(Wih, wih16, nullptr,
                                                       2LL * G4, inDim, Kpad);
  long long wtot2 = 2LL * G4 * HH;
  cast_f16_kernel<<<cdiv(wtot2, 256), 256, 0, stream>>>(Whh, whh16, nullptr,
                                                        2LL * G4, HH, HH);
  dim3 grid(G4 / 128, (unsigned)(rows / 32), 2);
  wmma_gemm_kernel<<<grid, 256, 0, stream>>>(
      x16, wih16, bias, xg,
      (int)rows, G4, Kpad, Kpad, Kpad, G4,
      /*sA=*/0, /*sW=*/(long long)G4 * Kpad, /*sD=*/rows * G4, /*sBias=*/G4);
  int Bb = (int)(rows / T);
  lstm_scan_kernel<<<dim3(Bb / 16, 2), 256, SCAN_LDS_BYTES, stream>>>(
      whh16, xg, hs_out, Bb, T);
}

extern "C" void kernel_launch(void* const* d_in, const int* in_sizes, int n_in,
                              void* d_out, int out_size, void* d_ws, size_t ws_size,
                              hipStream_t stream)
{
  const int*   p_tok = (const int*)d_in[0];
  const int*   q_tok = (const int*)d_in[1];
  const float* emb   = (const float*)d_in[2];
  const float* qenc_Wih = (const float*)d_in[3];
  const float* qenc_Whh = (const float*)d_in[4];
  const float* qenc_b   = (const float*)d_in[5];
  const float* penc_Wih = (const float*)d_in[6];
  const float* penc_Whh = (const float*)d_in[7];
  const float* penc_b   = (const float*)d_in[8];
  const float* m1_Wih = (const float*)d_in[9];
  const float* m1_Whh = (const float*)d_in[10];
  const float* m1_b   = (const float*)d_in[11];
  const float* m2_Wih = (const float*)d_in[12];
  const float* m2_Whh = (const float*)d_in[13];
  const float* m2_b   = (const float*)d_in[14];
  const float* out_Wih = (const float*)d_in[15];
  const float* out_Whh = (const float*)d_in[16];
  const float* out_b   = (const float*)d_in[17];
  const float* att_wc  = (const float*)d_in[18];
  const float* att_wq  = (const float*)d_in[19];
  const float* att_wcq = (const float*)d_in[20];
  const float* att_b   = (const float*)d_in[21];
  const float* p1_wg = (const float*)d_in[22];
  const float* p1_wm = (const float*)d_in[23];
  const float* p1_b  = (const float*)d_in[24];
  const float* p2_wg = (const float*)d_in[25];
  const float* p2_wm = (const float*)d_in[26];
  const float* p2_b  = (const float*)d_in[27];
  float* out = (float*)d_out;

  // ---- workspace carve (deterministic bump allocator) ----
  char* ws = (char*)d_ws;
  size_t pos = 0;
  auto alloc = [&](size_t bytes) -> void* {
    void* r = ws + pos;
    pos += (bytes + 255) & ~(size_t)255;
    return r;
  };
  const long long BC = (long long)BB * CC;      // 32768
  const long long BQ = (long long)BB * QQ;      // 4096
  _Float16* pX16   = (_Float16*)alloc(BC * EPAD * 2);
  _Float16* qX16   = (_Float16*)alloc(BQ * EPAD * 2);
  _Float16* wih16  = (_Float16*)alloc(2LL * G4 * G8 * 2);   // max Kpad=1024
  _Float16* whh16  = (_Float16*)alloc(2LL * G4 * HH * 2);
  float*    xg     = (float*)alloc(2LL * BC * G4 * 4);
  float*    qo     = (float*)alloc(BQ * D2 * 4);
  float*    co     = (float*)alloc(BC * D2 * 4);
  _Float16* x16    = (_Float16*)alloc(BC * G8 * 2);         // casts / g16
  float*    sbuf   = (float*)alloc(BC * QQ * 4);
  _Float16* a16    = (_Float16*)alloc(BC * QQ * 2);
  float*    smax   = (float*)alloc(BC * 4);
  float*    bw     = (float*)alloc(BC * 4);
  float*    cwc    = (float*)alloc(BC * 4);
  float*    qwq    = (float*)alloc(BQ * 4);
  float*    q2c    = (float*)alloc((long long)BB * D2 * 4);
  _Float16* qo16   = (_Float16*)alloc(BQ * D2 * 2);
  _Float16* qoT16  = (_Float16*)alloc((long long)BB * D2 * QQ * 2);
  _Float16* cosc16 = (_Float16*)alloc(BC * D2 * 2);
  float*    c2q    = (float*)alloc(BC * D2 * 4);
  float*    g32    = (float*)alloc(BC * G8 * 4);
  float*    hs1    = (float*)alloc(BC * D2 * 4);            // bilstm(g)
  float*    mbuf   = (float*)alloc(BC * D2 * 4);            // m
  float*    m2buf  = (float*)alloc(BC * D2 * 4);            // m2
  float*    pl1    = (float*)alloc(BC * 4);
  float*    pl2    = (float*)alloc(BC * 4);
  (void)ws_size; (void)in_sizes; (void)n_in; (void)out_size;

  // ---- 1. embedding gather + f16 cast (zero-padded to EPAD) ----
  gather_emb_kernel<<<cdiv(BC * EPAD, 256), 256, 0, stream>>>(p_tok, emb, pX16, BC);
  gather_emb_kernel<<<cdiv(BQ * EPAD, 256), 256, 0, stream>>>(q_tok, emb, qX16, BQ);

  // ---- 2. encoders ----
  run_bilstm(stream, qX16, BQ, QQ, EPAD, qenc_Wih, EE, qenc_Whh, qenc_b,
             wih16, whh16, xg, qo);
  run_bilstm(stream, pX16, BC, CC, EPAD, penc_Wih, EE, penc_Whh, penc_b,
             wih16, whh16, xg, co);

  // ---- 3. trilinear attention ----
  rowdot_kernel<<<cdiv(BC, 256), 256, 0, stream>>>(co, att_wc, cwc, BC, D2);
  rowdot_kernel<<<cdiv(BQ, 256), 256, 0, stream>>>(qo, att_wq, qwq, BQ, D2);
  cast_f16_kernel<<<cdiv(BQ * D2, 256), 256, 0, stream>>>(qo, qo16, nullptr, BQ, D2, D2);
  cast_f16_kernel<<<cdiv(BC * D2, 256), 256, 0, stream>>>(co, cosc16, att_wcq, BC, D2, D2);
  {   // s[b,c,q] = (co*wcq)[b,c,:] . qo[b,q,:]
    dim3 grid(1, CC / 32, BB);
    wmma_gemm_kernel<<<grid, 256, 0, stream>>>(
        cosc16, qo16, nullptr, sbuf, CC, QQ, D2, D2, D2, QQ,
        (long long)CC * D2, (long long)QQ * D2, (long long)CC * QQ, 0);
  }
  softmax_q_kernel<<<(unsigned)BC, 32, 0, stream>>>(sbuf, cwc, qwq, att_b, a16, smax);
  softmax_c_kernel<<<BB, 256, 0, stream>>>(smax, bw);
  q2c_kernel<<<cdiv((long long)BB * D2, 256), 256, 0, stream>>>(bw, co, q2c);
  qoT_kernel<<<cdiv((long long)BB * D2 * QQ, 256), 256, 0, stream>>>(qo, qoT16);
  {   // c2q[b,c,d] = a[b,c,:] . qo[b,:,d]
    dim3 grid(2, CC / 32, BB);
    wmma_gemm_kernel<<<grid, 256, 0, stream>>>(
        a16, qoT16, nullptr, c2q, CC, D2, QQ, QQ, QQ, D2,
        (long long)CC * QQ, (long long)D2 * QQ, (long long)CC * D2, 0);
  }
  build_g_kernel<<<cdiv(BC * D2, 256), 256, 0, stream>>>(co, c2q, q2c, g32, x16);

  // ---- 4. modeling stack ----
  run_bilstm(stream, x16, BC, CC, G8, m1_Wih, G8, m1_Whh, m1_b,
             wih16, whh16, xg, hs1);
  cast_f16_kernel<<<cdiv(BC * D2, 256), 256, 0, stream>>>(hs1, x16, nullptr, BC, D2, D2);
  run_bilstm(stream, x16, BC, CC, D2, m2_Wih, D2, m2_Whh, m2_b,
             wih16, whh16, xg, mbuf);
  cast_f16_kernel<<<cdiv(BC * D2, 256), 256, 0, stream>>>(mbuf, x16, nullptr, BC, D2, D2);
  run_bilstm(stream, x16, BC, CC, D2, out_Wih, D2, out_Whh, out_b,
             wih16, whh16, xg, m2buf);

  // ---- 5. pointer logits + masked softmax ----
  logits_kernel<<<cdiv(BC, 256), 256, 0, stream>>>(g32, mbuf, p1_wg, p1_wm, p1_b,
                                                   pl1, G8, D2);
  logits_kernel<<<cdiv(BC, 256), 256, 0, stream>>>(g32, m2buf, p2_wg, p2_wm, p2_b,
                                                   pl2, G8, D2);
  masked_softmax_kernel<<<BB, 256, 0, stream>>>(pl1, p_tok, out);
  masked_softmax_kernel<<<BB, 256, 0, stream>>>(pl2, p_tok, out + BC);
}
